// CrossNetMix2_48163763257625
// MI455X (gfx1250) — compile-verified, matrix operand order checked by reading, not run on hardware
//
#include <hip/hip_runtime.h>

typedef __attribute__((ext_vector_type(16))) __bf16 v16bf;
typedef __attribute__((ext_vector_type(8)))  __bf16 v8bf;
typedef __attribute__((ext_vector_type(8)))  float  v8f;

#define DDIM 512
#define NEXP 4
#define RDIM 32
#define NLAYER 2
#define ROWS_PER_BLK 16

// LDS row strides (elements), padded to kill bank conflicts on frag loads.
#define XF_STRIDE  516   // floats:  bank step 516 mod 64 = 4 per row
#define XBF_STRIDE 520   // bf16:    260 dwords/row -> lane bank = 4*mrow
#define RSTRIDE    40    // bf16:    20 dwords/row  -> lane bank = 20*mrow mod 64

static __device__ __forceinline__ float fast_tanh(float x) {
#if __has_builtin(__builtin_amdgcn_tanhf)
  return __builtin_amdgcn_tanhf(x);
#else
  float r;
  asm volatile("v_tanh_f32 %0, %1\n\tv_nop" : "=v"(r) : "v"(x));
  return r;
#endif
}

static __device__ __forceinline__ unsigned bfpack(float a, float b) {
  const unsigned short ua = __builtin_bit_cast(unsigned short, (__bf16)a);
  const unsigned short ub = __builtin_bit_cast(unsigned short, (__bf16)b);
  return (unsigned)ua | ((unsigned)ub << 16);
}

static __device__ __forceinline__ v16bf cat16(v8bf lo, v8bf hi) {
  v16bf r;
#pragma unroll
  for (int i = 0; i < 8; ++i) { r[i] = lo[i]; r[i + 8] = hi[i]; }
  return r;
}

static __device__ __forceinline__ v8f wmma_bf16(v16bf a, v16bf b, v8f c) {
  return __builtin_amdgcn_wmma_f32_16x16x32_bf16(false, a, false, b, (short)0, c,
                                                 false, false);
}

// GEMM-1 K-loop, specialized so the gate-GEMM variant has no divergent
// control flow inside the loop (keeps the scheduler free to pipeline loads).
template <bool WITH_GATE>
static __device__ __forceinline__ void
gemm1_loop(const __bf16* __restrict__ x_bf, const __bf16* __restrict__ Vte,
           const __bf16* __restrict__ Wgt, int mrow, int grp,
           v8f& acc0, v8f& acc1, v8f& accG) {
#pragma unroll 4
  for (int kc = 0; kc < DDIM; kc += 32) {
    const int klo = kc + grp * 8;
    const int khi = kc + 16 + grp * 8;
    const v16bf A = cat16(*(const v8bf*)&x_bf[mrow * XBF_STRIDE + klo],
                          *(const v8bf*)&x_bf[mrow * XBF_STRIDE + khi]);
    const v16bf B0 = cat16(*(const v8bf*)(Vte + mrow * DDIM + klo),
                           *(const v8bf*)(Vte + mrow * DDIM + khi));
    const v16bf B1 = cat16(*(const v8bf*)(Vte + (mrow + 16) * DDIM + klo),
                           *(const v8bf*)(Vte + (mrow + 16) * DDIM + khi));
    acc0 = wmma_bf16(A, B0, acc0);
    acc1 = wmma_bf16(A, B1, acc1);
    if constexpr (WITH_GATE) {
      const v16bf Bg = cat16(*(const v8bf*)(Wgt + mrow * DDIM + klo),
                             *(const v8bf*)(Wgt + mrow * DDIM + khi));
      accG = wmma_bf16(A, Bg, accG);
    }
  }
}

// ---------------------------------------------------------------------------
// Weight prep: f32 -> bf16.
//   Ub : [l][e][d][r]  (native layout; B-frag friendly for w @ U^T)
//   Vt : [l][e][r][d]  (transposed; B-frag friendly for x @ V)
//   Cb : [l][e][i][j]  (native; B-frag friendly for v @ C^T)
//   Wgt: [16][d]       (transposed + zero-padded n=4..15; B-frag for x @ Wg)
// ---------------------------------------------------------------------------
__global__ void prep_weights(const float* __restrict__ U,
                             const float* __restrict__ V,
                             const float* __restrict__ C,
                             const float* __restrict__ Wg,
                             __bf16* __restrict__ Ub,
                             __bf16* __restrict__ Vt,
                             __bf16* __restrict__ Cb,
                             __bf16* __restrict__ Wgt) {
  const int i = blockIdx.x * blockDim.x + threadIdx.x;
  const int nUV = NLAYER * NEXP * DDIM * RDIM;  // 131072
  if (i < nUV) {
    Ub[i] = (__bf16)U[i];
    const int le  = i / (RDIM * DDIM);
    const int rem = i % (RDIM * DDIM);
    const int r = rem / DDIM;
    const int d = rem % DDIM;
    Vt[i] = (__bf16)V[(le * DDIM + d) * RDIM + r];
  }
  if (i < NLAYER * NEXP * RDIM * RDIM) Cb[i] = (__bf16)C[i];
  if (i < 16 * DDIM) {
    const int n = i / DDIM, d = i % DDIM;
    Wgt[i] = (n < NEXP) ? (__bf16)Wg[d * NEXP + n] : (__bf16)0.0f;
  }
}

// ---------------------------------------------------------------------------
// One CrossNetMix layer. Block = 128 threads (4 waves) = one 16-row tile.
// Waves act as experts for the V/C GEMMs (wave 0 also folds the gate GEMM
// into its K-loop), then as d-range owners for the U GEMM + gated combine.
//   moe = x0 * (sum_e gate_e*uv_e + bias)   (softmax weights sum to 1)
// Safe with x_out == x_in (each block reads its rows before writing).
// ---------------------------------------------------------------------------
__global__ __launch_bounds__(128)
void crossnet_layer(const float* __restrict__ x_in,
                    const float* __restrict__ x0,
                    const __bf16* __restrict__ Vt,    // [E][R][D] bf16
                    const __bf16* __restrict__ Ub,    // [E][D][R] bf16
                    const __bf16* __restrict__ Cb,    // [E][R][R] bf16
                    const __bf16* __restrict__ Wgt,   // [16][D] bf16
                    const float* __restrict__ bias_l, // [D] f32
                    float* __restrict__ x_out) {
  __shared__ __align__(16) __bf16 x_bf[ROWS_PER_BLK * XBF_STRIDE];   // 16.3 KB
  __shared__ __align__(16) float  x_f [ROWS_PER_BLK * XF_STRIDE];    // 32.3 KB
  __shared__ float  logit_s[ROWS_PER_BLK * NEXP];
  __shared__ float  gate_s[NEXP * ROWS_PER_BLK];                     // [e][m]
  __shared__ __align__(16) __bf16 vx_s[NEXP * ROWS_PER_BLK * RSTRIDE]; // 5 KB
  __shared__ __align__(16) __bf16 w_s [NEXP * ROWS_PER_BLK * RSTRIDE]; // 5 KB

  const int t    = threadIdx.x;
  const int wave = t >> 5;       // 0..3: expert id, later d-chunk id
  const int lane = t & 31;
  const int mrow = lane & 15;    // n (B/C/D frags) or m (A frags)
  const int grp  = lane >> 4;    // 0/1: lane-half -> K-group select
  const int row0 = blockIdx.x * ROWS_PER_BLK;

  // ---- Stage: x tile (f32 + bf16, padded rows) into LDS -------------------
  {
    const float4* xsrc = (const float4*)(x_in + (size_t)row0 * DDIM);
    for (int i = t; i < ROWS_PER_BLK * (DDIM / 4); i += 128) {
      const int r = i >> 7, c4 = i & 127;          // 128 float4 per row
      const float4 v = xsrc[i];
      ((float4*)(x_f + r * XF_STRIDE))[c4] = v;
      uint2 p;
      p.x = bfpack(v.x, v.y);
      p.y = bfpack(v.z, v.w);
      *(uint2*)(x_bf + r * XBF_STRIDE + c4 * 4) = p;
    }
  }
  __syncthreads();

  // ---- Expert GEMM 1: v_x[e] = tanh(x @ V_e); wave 0 also does x @ Wg -----
  {
    const int e = wave;
    const __bf16* Vte = Vt + (size_t)e * RDIM * DDIM;
    v8f acc0 = {}, acc1 = {}, accG = {};
    if (wave == 0)
      gemm1_loop<true>(x_bf, Vte, Wgt, mrow, grp, acc0, acc1, accG);
    else
      gemm1_loop<false>(x_bf, Vte, Wgt, mrow, grp, acc0, acc1, accG);

    __bf16* vx = &vx_s[e * ROWS_PER_BLK * RSTRIDE];
#pragma unroll
    for (int i = 0; i < 8; ++i) {
      const int m = grp * 8 + i;   // D-matrix: VGPR i -> M = i + 8*(lane/16)
      vx[m * RSTRIDE + mrow]      = (__bf16)fast_tanh(acc0[i]);
      vx[m * RSTRIDE + mrow + 16] = (__bf16)fast_tanh(acc1[i]);
    }
    // ---- Gate softmax, entirely inside wave 0 ----------------------------
    if (wave == 0) {
      if (mrow < NEXP) {
#pragma unroll
        for (int i = 0; i < 8; ++i)
          logit_s[(grp * 8 + i) * NEXP + mrow] = accG[i];
      }
      __builtin_amdgcn_wave_barrier();
      if (lane < ROWS_PER_BLK) {
        const float l0 = logit_s[lane * NEXP + 0], l1 = logit_s[lane * NEXP + 1];
        const float l2 = logit_s[lane * NEXP + 2], l3 = logit_s[lane * NEXP + 3];
        const float mx = fmaxf(fmaxf(l0, l1), fmaxf(l2, l3));
        const float e0 = __expf(l0 - mx), e1 = __expf(l1 - mx);
        const float e2 = __expf(l2 - mx), e3 = __expf(l3 - mx);
        const float inv = 1.f / (e0 + e1 + e2 + e3);
        gate_s[0 * 16 + lane] = e0 * inv; gate_s[1 * 16 + lane] = e1 * inv;
        gate_s[2 * 16 + lane] = e2 * inv; gate_s[3 * 16 + lane] = e3 * inv;
      }
    }
  }
  __builtin_amdgcn_wave_barrier();  // wave-private LDS RAW ordering (vx_s)

  // ---- Expert GEMM 2: w[e] = tanh(v_x @ C_e^T)  (16x32 @ 32x32) -----------
  {
    const int e = wave;
    const __bf16* vx = &vx_s[e * ROWS_PER_BLK * RSTRIDE];
    const __bf16* Ce = Cb + e * RDIM * RDIM;
    const v16bf A = cat16(*(const v8bf*)&vx[mrow * RSTRIDE + grp * 8],
                          *(const v8bf*)&vx[mrow * RSTRIDE + 16 + grp * 8]);
    const v16bf B0 = cat16(*(const v8bf*)(Ce + mrow * RDIM + grp * 8),
                           *(const v8bf*)(Ce + mrow * RDIM + 16 + grp * 8));
    const v16bf B1 = cat16(*(const v8bf*)(Ce + (mrow + 16) * RDIM + grp * 8),
                           *(const v8bf*)(Ce + (mrow + 16) * RDIM + 16 + grp * 8));
    v8f c0 = {}, c1 = {};
    c0 = wmma_bf16(A, B0, c0);
    c1 = wmma_bf16(A, B1, c1);
    __bf16* wp = &w_s[e * ROWS_PER_BLK * RSTRIDE];
#pragma unroll
    for (int i = 0; i < 8; ++i) {
      const int m = grp * 8 + i;
      wp[m * RSTRIDE + mrow]      = (__bf16)fast_tanh(c0[i]);
      wp[m * RSTRIDE + mrow + 16] = (__bf16)fast_tanh(c1[i]);
    }
  }
  __syncthreads();  // all experts' w + gate visible to every wave

  // ---- U GEMM + gated combine: wave owns d in [128*wave, 128*wave+128) ----
  {
    v16bf Aw[NEXP];
    float g[NEXP][8];
#pragma unroll
    for (int e = 0; e < NEXP; ++e) {
      const __bf16* wp = &w_s[e * ROWS_PER_BLK * RSTRIDE];
      Aw[e] = cat16(*(const v8bf*)&wp[mrow * RSTRIDE + grp * 8],
                    *(const v8bf*)&wp[mrow * RSTRIDE + 16 + grp * 8]);
#pragma unroll
      for (int i = 0; i < 8; ++i) g[e][i] = gate_s[e * 16 + grp * 8 + i];
    }
    const int dbase = wave * (DDIM / 4);
#pragma unroll 2
    for (int nt = 0; nt < 8; ++nt) {
      const int dcol = dbase + nt * 16 + mrow;  // n = lane%16 within N-tile
      float sum[8] = {0.f, 0.f, 0.f, 0.f, 0.f, 0.f, 0.f, 0.f};
#pragma unroll
      for (int e = 0; e < NEXP; ++e) {
        const __bf16* ucol = Ub + ((size_t)e * DDIM + dcol) * RDIM;
        const v16bf B = cat16(*(const v8bf*)(ucol + grp * 8),
                              *(const v8bf*)(ucol + 16 + grp * 8));
        v8f tz = {};
        tz = wmma_bf16(Aw[e], B, tz);
#pragma unroll
        for (int i = 0; i < 8; ++i) sum[i] += g[e][i] * tz[i];
      }
      const float bv = bias_l[dcol];
#pragma unroll
      for (int i = 0; i < 8; ++i) {
        const int m = grp * 8 + i;
        const float xl  = x_f[m * XF_STRIDE + dcol];
        const float x0v = x0[(size_t)(row0 + m) * DDIM + dcol];
        x_out[(size_t)(row0 + m) * DDIM + dcol] = xl + x0v * (sum[i] + bv);
      }
    }
  }
}

// ---------------------------------------------------------------------------
extern "C" void kernel_launch(void* const* d_in, const int* in_sizes, int n_in,
                              void* d_out, int out_size, void* d_ws, size_t ws_size,
                              hipStream_t stream) {
  (void)in_sizes; (void)n_in; (void)out_size; (void)ws_size;
  const float* x    = (const float*)d_in[0];
  const float* U    = (const float*)d_in[1];
  const float* V    = (const float*)d_in[2];
  const float* C    = (const float*)d_in[3];
  const float* bias = (const float*)d_in[4];
  const float* Wg   = (const float*)d_in[5];
  float* out = (float*)d_out;

  const int nUV = NLAYER * NEXP * DDIM * RDIM;          // 131072
  const int nC  = NLAYER * NEXP * RDIM * RDIM;          // 8192
  __bf16* Ub  = (__bf16*)d_ws;
  __bf16* Vt  = Ub + nUV;
  __bf16* Cb  = Vt + nUV;
  __bf16* Wgt = Cb + nC;                                // 16*512

  prep_weights<<<(nUV + 255) / 256, 256, 0, stream>>>(U, V, C, Wg, Ub, Vt, Cb, Wgt);

  const int nblk = 16384 / ROWS_PER_BLK;  // 1024
  for (int l = 0; l < NLAYER; ++l) {
    const float* xin = (l == 0) ? x : out;  // layer 2 runs in place on d_out
    crossnet_layer<<<nblk, 128, 0, stream>>>(
        xin, x,
        Vt + (size_t)l * NEXP * RDIM * DDIM,
        Ub + (size_t)l * NEXP * DDIM * RDIM,
        Cb + (size_t)l * NEXP * RDIM * RDIM,
        Wgt, bias + (size_t)l * DDIM, out);
  }
}
